// Attention_52536039965434
// MI455X (gfx1250) — matrix operand order repaired and checked by reading, not compiled, not run on hardware
//
#include <hip/hip_runtime.h>
#include <cmath>

// ---------------- CDNA5 wave32 WMMA types ----------------
typedef __attribute__((ext_vector_type(16))) __bf16 v16bf;
typedef __attribute__((ext_vector_type(8)))  __bf16 v8bf;
typedef __attribute__((ext_vector_type(8)))  float  v8f;

#define WMMA_BF16(A, B, C) \
  __builtin_amdgcn_wmma_f32_16x16x32_bf16(false, (A), false, (B), (short)0, (C), false, false)

// Problem constants
#define BDIM 2
#define SQ   2048
#define SK   2048
#define HDIM 1024
#define NH   16
#define HD   64
#define QKVW 3072  // 3*H

// ---------------- helpers ----------------
__device__ __forceinline__ v16bf ldpair(const __bf16* p0, const __bf16* p1) {
  union { v16bf v; v8bf h[2]; } u;
  u.h[0] = *reinterpret_cast<const v8bf*>(p0);
  u.h[1] = *reinterpret_cast<const v8bf*>(p1);
  return u.v;
}

__device__ __forceinline__ float gelu_f(float x) {
  float x3 = x * x * x;
  return 0.5f * x * (1.0f + tanhf(0.7978845608028654f * (x + 0.044715f * x3)));
}

// ---------------- f32 -> bf16 conversion ----------------
__global__ void cvt_f32_bf16(const float* __restrict__ src, __bf16* __restrict__ dst, int n) {
  int i = blockIdx.x * blockDim.x + threadIdx.x;
  if (i < n) dst[i] = (__bf16)src[i];
}

// ---------------- tiled WMMA GEMM (double-buffered LDS) ----------------
// C[M,N] = act( A[M,K] * W[K,N] + bias[N] )
// A is split: cols [0,K0) from A0, cols [K0,K) from A1 (both row stride lda).
// Block tile 128x128, K-step 32, 8 waves (4x2), wave tile 32x64.
#define BM 128
#define BN 128
#define BK 32
#define LDSS 40   // bf16 elements per LDS row (80B, multiple of 16B)

template <bool GELU_F32>
__global__ __launch_bounds__(256)
void gemm_wmma_kernel(const __bf16* __restrict__ A0, const __bf16* __restrict__ A1, int K0,
                      const __bf16* __restrict__ W, const float* __restrict__ bias,
                      __bf16* __restrict__ Cb, float* __restrict__ Cf,
                      int M, int N, int K, int lda, int ldw) {
  __shared__ __bf16 As[2][BM * LDSS];
  __shared__ __bf16 Bs[2][BN * LDSS];   // transposed: Bs[buf][n*LDSS + k]

  const int tid  = threadIdx.x;
  const int l    = tid & 31;
  const int wid  = tid >> 5;     // 0..7
  const int wm   = wid >> 1;     // 0..3
  const int wn   = wid & 1;      // 0..1
  const int l16  = l & 15;
  const int half = l >> 4;
  const int kb   = half * 8;

  const int bn0 = blockIdx.x * BN;
  const int bm0 = blockIdx.y * BM;

  // per-thread staging positions
  const int am    = tid >> 1;         // A row 0..127
  const int aseg  = tid & 1;          // 16-col segment
  const int wk    = tid >> 3;         // W k-row 0..31
  const int wnseg = (tid & 7) * 16;   // W n-segment 0..112

  auto aSrc = [&](int gk) -> const __bf16* {
    return (gk < K0) ? A0 + (size_t)(bm0 + am) * lda + gk
                     : A1 + (size_t)(bm0 + am) * lda + (gk - K0);
  };
  auto loadTile = [&](int k0, v8bf& a0, v8bf& a1, v8bf& w0, v8bf& w1) {
    const __bf16* as = aSrc(k0 + aseg * 16);
    a0 = *reinterpret_cast<const v8bf*>(as);
    a1 = *reinterpret_cast<const v8bf*>(as + 8);
    const __bf16* wsrc = W + (size_t)(k0 + wk) * ldw + bn0 + wnseg;
    w0 = *reinterpret_cast<const v8bf*>(wsrc);
    w1 = *reinterpret_cast<const v8bf*>(wsrc + 8);
  };
  auto commitTile = [&](int buf, v8bf a0, v8bf a1, v8bf w0, v8bf w1) {
    *reinterpret_cast<v8bf*>(&As[buf][am * LDSS + aseg * 16])     = a0;
    *reinterpret_cast<v8bf*>(&As[buf][am * LDSS + aseg * 16 + 8]) = a1;
#pragma unroll
    for (int j = 0; j < 8; ++j) Bs[buf][(wnseg + j) * LDSS + wk]     = w0[j];
#pragma unroll
    for (int j = 0; j < 8; ++j) Bs[buf][(wnseg + 8 + j) * LDSS + wk] = w1[j];
  };

  v8f acc[2][4];
#pragma unroll
  for (int i = 0; i < 2; ++i)
#pragma unroll
    for (int j = 0; j < 4; ++j) acc[i][j] = (v8f){};

  v8bf a0, a1, w0, w1;
  loadTile(0, a0, a1, w0, w1);
  commitTile(0, a0, a1, w0, w1);
  __syncthreads();

  const int iters = K / BK;
  for (int it = 0; it < iters; ++it) {
    const int  buf  = it & 1;
    const bool last = (it == iters - 1);

    // issue next tile's global loads before computing (hide latency behind WMMAs)
    if (!last) loadTile((it + 1) * BK, a0, a1, w0, w1);
    // prefetch two tiles ahead into cache (global_prefetch_b8)
    if (it + 2 < iters) {
      const int pk = (it + 2) * BK;
      __builtin_prefetch(aSrc(pk + aseg * 16), 0, 0);
      __builtin_prefetch(W + (size_t)(pk + wk) * ldw + bn0 + wnseg, 0, 0);
    }

    // ---- compute from current buffer: 2 M-frags x 4 N-frags ----
#pragma unroll
    for (int mf = 0; mf < 2; ++mf) {
      const int row = wm * 32 + mf * 16 + l16;
      v16bf afr = ldpair(&As[buf][row * LDSS + kb], &As[buf][row * LDSS + 16 + kb]);
#pragma unroll
      for (int nf = 0; nf < 4; ++nf) {
        const int n = wn * 64 + nf * 16 + l16;
        v16bf bfr = ldpair(&Bs[buf][n * LDSS + kb], &Bs[buf][n * LDSS + 16 + kb]);
        acc[mf][nf] = WMMA_BF16(afr, bfr, acc[mf][nf]);
      }
    }

    if (!last) commitTile(buf ^ 1, a0, a1, w0, w1);
    __syncthreads();
  }

  // ---- epilogue: bias (+gelu) + store ----
#pragma unroll
  for (int mf = 0; mf < 2; ++mf) {
#pragma unroll
    for (int nf = 0; nf < 4; ++nf) {
      const int n    = bn0 + wn * 64 + nf * 16 + l16;
      const float bv = bias[n];
      const int mrow = bm0 + wm * 32 + mf * 16 + half * 8;
#pragma unroll
      for (int r = 0; r < 8; ++r) {
        float v = acc[mf][nf][r] + bv;
        if (GELU_F32) {
          Cf[(size_t)(mrow + r) * N + n] = gelu_f(v);
        } else {
          Cb[(size_t)(mrow + r) * N + n] = (__bf16)v;
        }
      }
    }
  }
}

// ---------------- flash attention (one wave per 16 q-rows of one head) ----------------
__global__ __launch_bounds__(128)
void attn_kernel(const __bf16* __restrict__ qkv_q,  // (B*SQ, 3H) bf16
                 const __bf16* __restrict__ qkv_k,  // (B*SK, 3H) bf16
                 const int*    __restrict__ mask,   // (B, SK)
                 __bf16*       __restrict__ out) {  // (B*SQ, H) bf16
  __shared__ __bf16 Pst[4][16 * LDSS];
  __shared__ __bf16 Vst[4][64 * LDSS];

  const int tid  = threadIdx.x;
  const int l    = tid & 31;
  const int w    = tid >> 5;
  const int l16  = l & 15;
  const int half = l >> 4;
  const int kb   = half * 8;

  const int gw  = blockIdx.x * 4 + w;
  const int b   = gw / (NH * (SQ / 16));
  const int rem = gw % (NH * (SQ / 16));
  const int h   = rem / (SQ / 16);
  const int q0  = (rem % (SQ / 16)) * 16;

  __bf16* P = Pst[w];
  __bf16* V = Vst[w];

  // Q A-fragments for d=[0,32) and d=[32,64)
  const __bf16* qbase = qkv_q + (size_t)(b * SQ + q0 + l16) * QKVW + h * HD;
  v16bf aq0 = ldpair(qbase + half * 16,      qbase + half * 16 + 8);
  v16bf aq1 = ldpair(qbase + 32 + half * 16, qbase + 40 + half * 16);

  float mrow[8], lrow[8];
#pragma unroll
  for (int r = 0; r < 8; ++r) { mrow[r] = -1.0e30f; lrow[r] = 0.0f; }
  v8f accv[4];
#pragma unroll
  for (int g = 0; g < 4; ++g) accv[g] = (v8f){};

  const __bf16* kbase = qkv_k + (size_t)(b * SK) * QKVW + HDIM + h * HD;
  const __bf16* vbase = qkv_k + (size_t)(b * SK) * QKVW + 2 * HDIM + h * HD;
  const int* mb = mask + b * SK;

  for (int k0 = 0; k0 < SK; k0 += 32) {
    // prefetch next chunk's K and V rows while this chunk computes
    if (k0 + 32 < SK) {
      __builtin_prefetch(vbase + (size_t)(k0 + 32 + l) * QKVW, 0, 0);
      __builtin_prefetch(kbase + (size_t)(k0 + 32 + l) * QKVW, 0, 0);
    }

    // ---- stage V chunk (32 x 64) transposed into LDS: Vst[d][k] ----
    {
      const __bf16* vsrc = vbase + (size_t)(k0 + l) * QKVW;
      v8bf t[8];
#pragma unroll
      for (int j = 0; j < 8; ++j) t[j] = *reinterpret_cast<const v8bf*>(vsrc + j * 8);
#pragma unroll
      for (int j = 0; j < 8; ++j)
#pragma unroll
        for (int e = 0; e < 8; ++e) V[(j * 8 + e) * LDSS + l] = t[j][e];
    }

    // ---- scores: two 16x16 tiles (k cols k0..k0+31) ----
    v8f s[2];
#pragma unroll
    for (int t = 0; t < 2; ++t) {
      const __bf16* kc = kbase + (size_t)(k0 + t * 16 + l16) * QKVW + half * 16;
      v16bf bk0 = ldpair(kc,      kc + 8);       // d chunk [0,32)
      v16bf bk1 = ldpair(kc + 32, kc + 40);      // d chunk [32,64)
      v8f sc = (v8f){};
      sc = WMMA_BF16(aq0, bk0, sc);
      sc = WMMA_BF16(aq1, bk1, sc);
      s[t] = sc;
    }

    // ---- multiplicative mask (reference: scores * mask, then softmax) ----
    const float mk0 = (float)mb[k0 + l16];
    const float mk1 = (float)mb[k0 + 16 + l16];
#pragma unroll
    for (int r = 0; r < 8; ++r) { s[0][r] *= mk0; s[1][r] *= mk1; }

    // ---- online softmax: row reductions across 16 lanes ----
    float cmax[8];
#pragma unroll
    for (int r = 0; r < 8; ++r) cmax[r] = fmaxf(s[0][r], s[1][r]);
#pragma unroll
    for (int off = 1; off < 16; off <<= 1)
#pragma unroll
      for (int r = 0; r < 8; ++r) cmax[r] = fmaxf(cmax[r], __shfl_xor(cmax[r], off, 32));

    float scale[8];
#pragma unroll
    for (int r = 0; r < 8; ++r) {
      float mn = fmaxf(mrow[r], cmax[r]);
      scale[r] = __expf(mrow[r] - mn);
      mrow[r]  = mn;
    }
    float csum[8];
#pragma unroll
    for (int r = 0; r < 8; ++r) {
      float p0 = __expf(s[0][r] - mrow[r]);
      float p1 = __expf(s[1][r] - mrow[r]);
      s[0][r] = p0; s[1][r] = p1;
      csum[r] = p0 + p1;
    }
#pragma unroll
    for (int off = 1; off < 16; off <<= 1)
#pragma unroll
      for (int r = 0; r < 8; ++r) csum[r] += __shfl_xor(csum[r], off, 32);
#pragma unroll
    for (int r = 0; r < 8; ++r) lrow[r] = lrow[r] * scale[r] + csum[r];
#pragma unroll
    for (int g = 0; g < 4; ++g)
#pragma unroll
      for (int r = 0; r < 8; ++r) accv[g][r] *= scale[r];

    // ---- stage P (C layout -> LDS) for A-fragment reuse ----
#pragma unroll
    for (int t = 0; t < 2; ++t)
#pragma unroll
      for (int r = 0; r < 8; ++r)
        P[(r + half * 8) * LDSS + t * 16 + l16] = (__bf16)s[t][r];

    // intra-wave LDS RAW: no barrier needed, just drain DScnt
    asm volatile("s_wait_dscnt 0" ::: "memory");

    // ---- P @ V: A = P (16x32), B = V chunk columns ----
    v16bf ap = ldpair(&P[l16 * LDSS + half * 16], &P[l16 * LDSS + half * 16 + 8]);
#pragma unroll
    for (int g = 0; g < 4; ++g) {
      v16bf bv = ldpair(&V[(g * 16 + l16) * LDSS + half * 16],
                        &V[(g * 16 + l16) * LDSS + half * 16 + 8]);
      accv[g] = WMMA_BF16(ap, bv, accv[g]);
    }
  }

  // ---- normalize and store ----
  __bf16* obase = out + (size_t)(b * SQ + q0) * HDIM + h * HD;
#pragma unroll
  for (int g = 0; g < 4; ++g) {
#pragma unroll
    for (int r = 0; r < 8; ++r) {
      const int row = r + half * 8;
      const int d   = g * 16 + l16;
      obase[(size_t)row * HDIM + d] = (__bf16)(accv[g][r] / lrow[r]);
    }
  }
}

// ---------------- host orchestration ----------------
extern "C" void kernel_launch(void* const* d_in, const int* in_sizes, int n_in,
                              void* d_out, int out_size, void* d_ws, size_t ws_size,
                              hipStream_t stream) {
  const float* attender = (const float*)d_in[0];  // (B*SQ, H)
  const float* attendee = (const float*)d_in[1];  // (B*SK, H)
  const int*   mask     = (const int*)  d_in[2];  // (B, SK)
  const float* c_attn_w = (const float*)d_in[3];  // (H, 3H)
  const float* c_attn_b = (const float*)d_in[4];  // (3H)
  const float* c_proj_w = (const float*)d_in[5];  // (H, H)
  const float* c_proj_b = (const float*)d_in[6];  // (H)
  const float* mlp_w    = (const float*)d_in[7];  // (2H, H)
  const float* mlp_b    = (const float*)d_in[8];  // (H)
  float* out = (float*)d_out;                     // (B*SQ, H)

  const int MQ = BDIM * SQ;  // 4096 rows
  char* ws = (char*)d_ws;
  size_t off = 0;
  auto alloc = [&](size_t bytes) { char* p = ws + off; off += (bytes + 255) & ~(size_t)255; return p; };

  __bf16* Xq    = (__bf16*)alloc((size_t)MQ * HDIM * 2);
  __bf16* Xk    = (__bf16*)alloc((size_t)MQ * HDIM * 2);
  __bf16* Wattn = (__bf16*)alloc((size_t)HDIM * QKVW * 2);
  __bf16* Wproj = (__bf16*)alloc((size_t)HDIM * HDIM * 2);
  __bf16* Wmlp  = (__bf16*)alloc((size_t)2 * HDIM * HDIM * 2);
  __bf16* QKVq  = (__bf16*)alloc((size_t)MQ * QKVW * 2);
  __bf16* QKVk  = (__bf16*)alloc((size_t)MQ * QKVW * 2);
  __bf16* Attn  = (__bf16*)alloc((size_t)MQ * HDIM * 2);
  __bf16* Proj  = (__bf16*)alloc((size_t)MQ * HDIM * 2);

  auto cvt = [&](const float* s, __bf16* d, int n) {
    cvt_f32_bf16<<<(n + 255) / 256, 256, 0, stream>>>(s, d, n);
  };
  cvt(attender, Xq,    MQ * HDIM);
  cvt(attendee, Xk,    MQ * HDIM);
  cvt(c_attn_w, Wattn, HDIM * QKVW);
  cvt(c_proj_w, Wproj, HDIM * HDIM);
  cvt(mlp_w,    Wmlp,  2 * HDIM * HDIM);

  // QKV projections: (4096 x 1024) @ (1024 x 3072) + b
  gemm_wmma_kernel<false><<<dim3(QKVW / BN, MQ / BM), 256, 0, stream>>>(
      Xq, Xq, HDIM, Wattn, c_attn_b, QKVq, nullptr, MQ, QKVW, HDIM, HDIM, QKVW);
  gemm_wmma_kernel<false><<<dim3(QKVW / BN, MQ / BM), 256, 0, stream>>>(
      Xk, Xk, HDIM, Wattn, c_attn_b, QKVk, nullptr, MQ, QKVW, HDIM, HDIM, QKVW);

  // Attention: one wave per (b, h, 16-q-row tile) -> B*NH*(SQ/16)/4 blocks of 128
  attn_kernel<<<(BDIM * NH * (SQ / 16)) / 4, 128, 0, stream>>>(QKVq, QKVk, mask, Attn);

  // Output projection: (4096 x 1024) @ (1024 x 1024) + b
  gemm_wmma_kernel<false><<<dim3(HDIM / BN, MQ / BM), 256, 0, stream>>>(
      Attn, Attn, HDIM, Wproj, c_proj_b, Proj, nullptr, MQ, HDIM, HDIM, HDIM, HDIM);

  // MLP on concat [attender | proj]: (4096 x 2048) @ (2048 x 1024) + b, GELU, f32 out
  gemm_wmma_kernel<true><<<dim3(HDIM / BN, MQ / BM), 256, 0, stream>>>(
      Xq, Proj, HDIM, Wmlp, mlp_b, nullptr, out, MQ, HDIM, 2 * HDIM, HDIM, HDIM);
}
